// DRIFT_NET_12189117186091
// MI455X (gfx1250) — matrix operand order repaired and checked by reading, not compiled, not run on hardware
//
#include <hip/hip_runtime.h>

typedef _Float16 v8h  __attribute__((ext_vector_type(8)));
typedef _Float16 v16h __attribute__((ext_vector_type(16)));
typedef float    v8f  __attribute__((ext_vector_type(8)));
typedef float    v4f  __attribute__((ext_vector_type(4)));

union V16 { v16h v; v8h h[2]; };

#define KDIM 256
#define HW   6400
#define WDIM 80
#define NPIX 51200        // B*H*W
#define LDS_PITCH 264     // 256 + 8 halves pad: 528B row = 132 dwords, 132%64=4 -> conflict-free

static __device__ __forceinline__ v8f wmma_f16(v16h a, v16h b, v8f c) {
  // v_wmma_f32_16x16x32_f16  (neg_a, A, neg_b, B, c_mod, C, reuse_a, reuse_b)
  return __builtin_amdgcn_wmma_f32_16x16x32_f16(false, a, false, b, (short)0, c, false, false);
}

// CDNA5 async copy: memory -> LDS without touching VGPRs (tracked by ASYNCcnt).
static __device__ __forceinline__ void async_copy_b128(unsigned lds_off, const void* gaddr) {
  asm volatile("global_load_async_to_lds_b128 %0, %1, off"
               :: "v"(lds_off), "v"(gaddr) : "memory");
}
static __device__ __forceinline__ void wait_asynccnt0() {
  asm volatile("s_wait_asynccnt 0x0" ::: "memory");
}

// ---------------------------------------------------------------------------
// Kernel 0: convert/concat weights to f16, concat biases.
// Wcat rows: [0,256)=Wv, [256,320)=Woff, [320,352)=Wa  (all K=256 row-major)
// ---------------------------------------------------------------------------
__global__ __launch_bounds__(256) void prep_kernel(
    const float* __restrict__ Wv, const float* __restrict__ Woff,
    const float* __restrict__ Wa, const float* __restrict__ Wo,
    const float* __restrict__ bv, const float* __restrict__ boff,
    const float* __restrict__ ba,
    _Float16* __restrict__ Wcat, _Float16* __restrict__ Wo16,
    float* __restrict__ bcat) {
  int i = blockIdx.x * 256 + threadIdx.x;
  if (i < 65536) {
    Wcat[i] = (_Float16)Wv[i];
  } else if (i < 81920) {
    Wcat[i] = (_Float16)Woff[i - 65536];
  } else if (i < 90112) {
    Wcat[i] = (_Float16)Wa[i - 81920];
  } else if (i < 155648) {
    Wo16[i - 90112] = (_Float16)Wo[i - 90112];
  } else if (i < 156000) {
    int j = i - 155648;
    bcat[j] = (j < 256) ? bv[j] : (j < 320) ? boff[j - 256] : ba[j - 320];
  }
}

// ---------------------------------------------------------------------------
// Kernel 1: x [B,256,6400] f32  ->  xT [B*6400, 256] f16 (LDS tile transpose)
// ---------------------------------------------------------------------------
__global__ __launch_bounds__(256) void transpose_kernel(
    const float* __restrict__ x, _Float16* __restrict__ xT) {
  __shared__ _Float16 tile[32][33];
  int tb = blockIdx.x;
  int pt = tb % 200;          // pixel tile
  int ct = (tb / 200) & 7;    // channel tile
  int b  = tb / 1600;
  int tx = threadIdx.x & 31;
  int ty = threadIdx.x >> 5;  // 0..7
  const float* src = x + ((size_t)(b * 256 + ct * 32)) * HW + pt * 32;
#pragma unroll
  for (int i = 0; i < 4; i++) {
    int c = ty + i * 8;
    tile[c][tx] = (_Float16)src[(size_t)c * HW + tx];
  }
  __syncthreads();
  _Float16* dst = xT + ((size_t)(b * HW + pt * 32)) * 256 + ct * 32;
#pragma unroll
  for (int i = 0; i < 4; i++) {
    int p = ty + i * 8;
    dst[(size_t)p * 256 + tx] = tile[tx][p];  // coalesced in channel
  }
}

// ---------------------------------------------------------------------------
// Stage a 64-pixel x 256-K f16 B tile into LDS with async copies.
// Per staging instruction: lane stride 16B -> 512B contiguous global reads;
// LDS writes 4 dwords/lane, conflict-free.
// ---------------------------------------------------------------------------
static __device__ __forceinline__ void stage_B(
    _Float16* Bs, const _Float16* __restrict__ Bm, int n0) {
  const int t = threadIdx.x;
  const int w = t >> 5, lane = t & 31;
  const _Float16* gsrc = Bm + (size_t)n0 * KDIM;  // 64 rows x 256 halves contiguous
#pragma unroll
  for (int i = 0; i < 8; i++) {
    int row = i * 8 + w;
    int col = lane * 8;
    unsigned lds_off = (unsigned)(uintptr_t)(&Bs[row * LDS_PITCH + col]);
    async_copy_b128(lds_off, gsrc + (size_t)row * KDIM + col);
  }
  wait_asynccnt0();   // wave's own async copies complete
  __syncthreads();    // tile visible to all 8 waves
}

// ---------------------------------------------------------------------------
// Per-wave 16(M) x 64(N) WMMA tile, K=256; A from global (L1-resident
// weights), B fragments from the shared LDS tile. All four B fragments of a
// K-chunk are loaded up front so the compiler can clause the 8 ds_load_b128,
// wait once, and issue the 4 WMMAs back-to-back.
// ---------------------------------------------------------------------------
static __device__ __forceinline__ void gemm_accum(
    const _Float16* __restrict__ A, const _Float16* Bs,
    int m0, int l15, int hi, v8f acc[4]) {
  const _Float16* Ap = A + (size_t)(m0 + l15) * KDIM + hi * 8;
  const _Float16* Bp = Bs + l15 * LDS_PITCH + hi * 8;
#pragma unroll
  for (int kc = 0; kc < KDIM; kc += 32) {
    V16 a;
    a.h[0] = *(const v8h*)(Ap + kc);
    a.h[1] = *(const v8h*)(Ap + kc + 16);
    V16 bf[4];
#pragma unroll
    for (int j = 0; j < 4; j++) {
      const _Float16* bp = Bp + j * (16 * LDS_PITCH) + kc;
      bf[j].h[0] = *(const v8h*)(bp);      // ds_load_b128
      bf[j].h[1] = *(const v8h*)(bp + 16);  // ds_load_b128
    }
#pragma unroll
    for (int j = 0; j < 4; j++) {
      acc[j] = wmma_f16(a.v, bf[j].v, acc[j]);
    }
  }
}

// ---------------------------------------------------------------------------
// Kernel 2: fused GEMM1: M=352 (value 256 | offsets 64 | attn 32), N=51200.
// Grid: 800 n-blocks x 3 m-groups (8 waves x 16 rows each; last group 6 live).
// Epilogue: value -> f16 [n][256]; offsets -> f32 [n][64];
//           attn  -> softmax over groups of 4 (in-lane), f32 [n][32].
// ---------------------------------------------------------------------------
__global__ __launch_bounds__(256) void gemm1_kernel(
    const _Float16* __restrict__ Wcat, const float* __restrict__ bcat,
    const _Float16* __restrict__ xT,
    _Float16* __restrict__ valueT, float* __restrict__ offs,
    float* __restrict__ attnW) {
  __shared__ _Float16 Bs[64 * LDS_PITCH];
  const int nb = blockIdx.x % 800;
  const int mg = blockIdx.x / 800;
  const int n0 = nb * 64;
  const int lane = threadIdx.x & 31;
  const int wv   = threadIdx.x >> 5;
  const int m_tile = mg * 8 + wv;
  const int l15 = lane & 15;
  const int hi  = lane >> 4;

  stage_B(Bs, xT, n0);          // all waves participate (barrier inside)

  if (m_tile >= 22) return;     // wave-uniform; EXEC all-ones for live waves
  const int m0 = m_tile * 16;

  v8f acc[4] = {};
  gemm_accum(Wcat, Bs, m0, l15, hi, acc);

  const int mrow = m0 + hi * 8;  // 8 contiguous output channels per lane
  float bias[8];
#pragma unroll
  for (int r = 0; r < 8; r++) bias[r] = bcat[mrow + r];

  if (m0 < 256) {                // ---- value channels ----
#pragma unroll
    for (int j = 0; j < 4; j++) {
      int n = n0 + j * 16 + l15;
      v8h o;
#pragma unroll
      for (int r = 0; r < 8; r++) o[r] = (_Float16)(acc[j][r] + bias[r]);
      *(v8h*)(valueT + (size_t)n * 256 + mrow) = o;
    }
  } else if (m0 < 320) {         // ---- offset channels ----
    int co = mrow - 256;
#pragma unroll
    for (int j = 0; j < 4; j++) {
      int n = n0 + j * 16 + l15;
      v4f o0, o1;
#pragma unroll
      for (int r = 0; r < 4; r++) {
        o0[r] = acc[j][r] + bias[r];
        o1[r] = acc[j][r + 4] + bias[r + 4];
      }
      float* dst = offs + (size_t)n * 64 + co;
      *(v4f*)dst = o0;
      *(v4f*)(dst + 4) = o1;
    }
  } else {                       // ---- attn: softmax over NP=4, in-lane ----
    int co = mrow - 320;         // multiple of 8 -> two full 4-point groups
#pragma unroll
    for (int j = 0; j < 4; j++) {
      int n = n0 + j * 16 + l15;
      float l[8];
#pragma unroll
      for (int r = 0; r < 8; r++) l[r] = acc[j][r] + bias[r];
#pragma unroll
      for (int g = 0; g < 8; g += 4) {
        float mx = fmaxf(fmaxf(l[g], l[g + 1]), fmaxf(l[g + 2], l[g + 3]));
        float e0 = __expf(l[g] - mx),     e1 = __expf(l[g + 1] - mx);
        float e2 = __expf(l[g + 2] - mx), e3 = __expf(l[g + 3] - mx);
        float inv = 1.0f / (e0 + e1 + e2 + e3);
        l[g] = e0 * inv; l[g + 1] = e1 * inv; l[g + 2] = e2 * inv; l[g + 3] = e3 * inv;
      }
      v4f o0, o1;
#pragma unroll
      for (int r = 0; r < 4; r++) { o0[r] = l[r]; o1[r] = l[r + 4]; }
      float* dst = attnW + (size_t)n * 32 + co;
      *(v4f*)dst = o0;
      *(v4f*)(dst + 4) = o1;
    }
  }
}

// ---------------------------------------------------------------------------
// Kernel 3: bilinear sampling + point-weighted sum.
// One wave per (pixel, head); lane = channel within head (CPH=32).
// value stays L2-resident (26 MB << 192 MB) so gathers are on-chip.
// ---------------------------------------------------------------------------
__global__ __launch_bounds__(256) void sample_kernel(
    const _Float16* __restrict__ valueT, const float* __restrict__ offs,
    const float* __restrict__ attnW, _Float16* __restrict__ wT) {
  int lane = threadIdx.x & 31;
  int gw = blockIdx.x * 8 + (threadIdx.x >> 5);  // 0 .. 409599
  int head = gw & 7;
  int n = gw >> 3;                               // 0 .. 51199
  int b = n / HW;
  int p = n % HW;
  int py = p / WDIM, px = p % WDIM;
  const float* op = offs  + (size_t)n * 64 + head * 8;   // [np*2 + {x,y}]
  const float* ap = attnW + (size_t)n * 32 + head * 4;
  const _Float16* vbase = valueT + (size_t)b * HW * 256 + head * 32 + lane;

  const float sc = 0.5f * 79.0f;                 // align_corners=True
  float gx = px * (2.0f / 79.0f) - 1.0f;
  float gy = py * (2.0f / 79.0f) - 1.0f;
  float accv = 0.0f;
#pragma unroll
  for (int q = 0; q < 4; q++) {
    float ix = (gx + op[q * 2 + 0] + 1.0f) * sc;
    float iy = (gy + op[q * 2 + 1] + 1.0f) * sc;
    float aw = ap[q];
    float x0f = floorf(ix), y0f = floorf(iy);
    int x0 = (int)x0f, y0 = (int)y0f;
    float wx1 = ix - x0f, wx0 = 1.0f - wx1;
    float wy1 = iy - y0f, wy0 = 1.0f - wy1;
    float s = 0.0f;
#pragma unroll
    for (int cy = 0; cy < 2; cy++) {
      int yy = y0 + cy;
      if (yy < 0 || yy > 79) continue;
      float wy = cy ? wy1 : wy0;
#pragma unroll
      for (int cx = 0; cx < 2; cx++) {
        int xx = x0 + cx;
        if (xx < 0 || xx > 79) continue;
        float wx = cx ? wx1 : wx0;
        float v = (float)vbase[((size_t)(yy * WDIM + xx)) * 256];  // 64B/wave, coalesced
        s += wx * wy * v;
      }
    }
    accv += aw * s;
  }
  wT[(size_t)n * 256 + head * 32 + lane] = (_Float16)accv;
}

// ---------------------------------------------------------------------------
// Kernel 4: GEMM2 (Wo @ weighted) + bias + residual, output [B,256,H,W] f32.
// Grid: 800 n-blocks x 2 m-groups (8 waves x 16 rows = 128 rows each).
// ---------------------------------------------------------------------------
__global__ __launch_bounds__(256) void gemm2_kernel(
    const _Float16* __restrict__ Wo16, const float* __restrict__ bo,
    const _Float16* __restrict__ wT, const float* __restrict__ xin,
    float* __restrict__ out) {
  __shared__ _Float16 Bs[64 * LDS_PITCH];
  const int nb = blockIdx.x % 800;
  const int mg = blockIdx.x / 800;
  const int n0 = nb * 64;
  const int lane = threadIdx.x & 31;
  const int wv   = threadIdx.x >> 5;
  const int m0 = (mg * 8 + wv) * 16;   // always < 256
  const int l15 = lane & 15;
  const int hi  = lane >> 4;

  stage_B(Bs, wT, n0);

  v8f acc[4] = {};
  gemm_accum(Wo16, Bs, m0, l15, hi, acc);

  const int mrow = m0 + hi * 8;
  float bias[8];
#pragma unroll
  for (int r = 0; r < 8; r++) bias[r] = bo[mrow + r];

#pragma unroll
  for (int j = 0; j < 4; j++) {
    int n = n0 + j * 16 + l15;
    int b = n / HW;
    int p = n % HW;
    size_t base = ((size_t)b * 256 + mrow) * HW + p;
#pragma unroll
    for (int r = 0; r < 8; r++) {
      out[base + (size_t)r * HW] = acc[j][r] + bias[r] + xin[base + (size_t)r * HW];
    }
  }
}

// ---------------------------------------------------------------------------
extern "C" void kernel_launch(void* const* d_in, const int* in_sizes, int n_in,
                              void* d_out, int out_size, void* d_ws, size_t ws_size,
                              hipStream_t stream) {
  const float* x    = (const float*)d_in[0];
  const float* Wv   = (const float*)d_in[1];
  const float* bv   = (const float*)d_in[2];
  const float* Woff = (const float*)d_in[3];
  const float* boff = (const float*)d_in[4];
  const float* Wa   = (const float*)d_in[5];
  const float* ba   = (const float*)d_in[6];
  const float* Wo   = (const float*)d_in[7];
  const float* bo   = (const float*)d_in[8];
  float* out = (float*)d_out;

  char* ws = (char*)d_ws;
  _Float16* xT     = (_Float16*)(ws);               // 51200*256*2 = 26,214,400
  _Float16* valueT = (_Float16*)(ws + 26214400);    // 26,214,400
  _Float16* wT     = (_Float16*)(ws + 52428800);    // 26,214,400
  float*    offs   = (float*)   (ws + 78643200);    // 51200*64*4 = 13,107,200
  float*    attnW  = (float*)   (ws + 91750400);    // 51200*32*4 =  6,553,600
  _Float16* Wcat   = (_Float16*)(ws + 98304000);    // 352*256*2  =    180,224
  _Float16* Wo16   = (_Float16*)(ws + 98484224);    // 256*256*2  =    131,072
  float*    bcat   = (float*)   (ws + 98615296);    // 352*4      =      1,408

  prep_kernel<<<610, 256, 0, stream>>>(Wv, Woff, Wa, Wo, bv, boff, ba,
                                       Wcat, Wo16, bcat);
  transpose_kernel<<<12800, 256, 0, stream>>>(x, xT);
  gemm1_kernel<<<2400, 256, 0, stream>>>(Wcat, bcat, xT, valueT, offs, attnW);
  sample_kernel<<<51200, 256, 0, stream>>>(valueT, offs, attnW, wT);
  gemm2_kernel<<<1600, 256, 0, stream>>>(Wo16, bo, wT, x, out);
}
